// SierpinskiTransformerLayer_1846835937858
// MI455X (gfx1250) — matrix operand order repaired and checked by reading, not compiled, not run on hardware
//
#include <hip/hip_runtime.h>
#include <hip/hip_bf16.h>
#include <math.h>

// ---------------------------------------------------------------------------
// SierpinskiTransformerLayer for MI455X (gfx1250, wave32, WMMA).
// All GEMMs run through a bf16 WMMA tile kernel (v_wmma_f32_16x16x32_bf16,
// f32 accumulate) with DOUBLE-BUFFERED LDS tiles filled by the CDNA5 async
// copy path (global_load_async_to_lds_b128 + s_wait_asynccnt), so the next
// K-tile streams into LDS while the current one feeds the matrix pipes.
// Attention is multi-pass (scores -> masked softmax -> PV), exploiting the
// 192MB L2 (score tensor = 96MB bf16 stays resident).
// ---------------------------------------------------------------------------

typedef __bf16 bf16_t;
typedef __bf16 v16bf __attribute__((ext_vector_type(16)));
typedef float  v8f   __attribute__((ext_vector_type(8)));

#define B_DIM 4
#define S_DIM 1024
#define E_DIM 768
#define H_DIM 12
#define D_DIM 64
#define F_DIM 3072

struct GemmParams {
    const bf16_t* A;     // [M,K] row-major bf16
    const bf16_t* B;     // [N,K] row-major bf16 (i.e. W^T)
    float*        outF;
    bf16_t*       outB;
    const float*  bias;  // [N] or nullptr
    int M, N, K;
    int lda, ldb, ldo;
    long long strideAz, strideBz, strideOz; // per-blockIdx.z element strides
    float scale;
};

union BfFrag { uint4 u[2]; v16bf v; };

// CDNA5 async global->LDS copy, 16B per lane, tracked by ASYNCcnt.
// ldsOff = LDS byte offset (low 32 bits of the generic shared address),
// gaddr  = 64-bit global address.
__device__ __forceinline__ void async_g2l_b128(unsigned ldsOff,
                                               const void* gaddr)
{
    asm volatile("global_load_async_to_lds_b128 %0, %1, off"
                 :
                 : "v"(ldsOff), "v"((unsigned long long)(uintptr_t)gaddr)
                 : "memory");
}

__device__ __forceinline__ void wait_asynccnt0()
{
    asm volatile("s_wait_asynccnt 0x0" ::: "memory");
}

__device__ __forceinline__ unsigned lds_offset_of(const void* p)
{
    return (unsigned)(uintptr_t)p;   // low 32 bits of flat shared address
}

// ---------------------------------------------------------------------------
// Epilogue modes:
//  0: f32 out, += bias                       (out-proj pre-LN, FFN2 pre-LN)
//  1: bf16 out, += bias, scatter [B,H,S,D]   (Q, K)
//  2: bf16 out, += bias, scatter [B,H,D,S]   (V transposed for PV GEMM)
//  3: bf16 out, *scale, Sierpinski mask      (attention scores, pre-softmax)
//  4: bf16 out, gather heads -> [B,S,E]      (attended)
//  5: bf16 out, += bias, exact GELU          (FFN1)
// ---------------------------------------------------------------------------
template <int MODE, int FM, int FN>
__device__ inline void gemm_epilogue(const GemmParams& p, v8f acc[FM][FN],
                                     int mBase, int nBase, int lhalf, int l15,
                                     int z)
{
#pragma unroll
    for (int fm = 0; fm < FM; ++fm) {
#pragma unroll
        for (int fn = 0; fn < FN; ++fn) {
#pragma unroll
            for (int v = 0; v < 8; ++v) {
                const int m = mBase + fm * 16 + lhalf * 8 + v;
                const int n = nBase + fn * 16 + l15;
                float val = acc[fm][fn][v];
                if constexpr (MODE == 0) {
                    val += p.bias[n];
                    p.outF[(size_t)m * p.ldo + n] = val;
                } else if constexpr (MODE == 1) {
                    val += p.bias[n];
                    const int b = m >> 10, s = m & 1023;
                    const int h = n >> 6,  d = n & 63;
                    p.outB[(((size_t)(b * H_DIM + h) << 10) + s) * D_DIM + d] =
                        (bf16_t)val;
                } else if constexpr (MODE == 2) {
                    val += p.bias[n];
                    const int b = m >> 10, s = m & 1023;
                    const int h = n >> 6,  d = n & 63;
                    p.outB[((size_t)(b * H_DIM + h) * D_DIM + d) * S_DIM + s] =
                        (bf16_t)val;
                } else if constexpr (MODE == 3) {
                    val *= p.scale;
                    if (m & n) val = 0.0f;   // Sierpinski: keep iff (q&k)==0
                    p.outB[(size_t)z * p.strideOz + (size_t)m * p.ldo + n] =
                        (bf16_t)val;
                } else if constexpr (MODE == 4) {
                    const int b = z / H_DIM, h = z % H_DIM;
                    p.outB[((size_t)(b << 10) + m) * E_DIM + h * D_DIM + n] =
                        (bf16_t)val;
                } else if constexpr (MODE == 5) {
                    val += p.bias[n];
                    val = 0.5f * val * (1.0f + erff(val * 0.70710678118654752f));
                    p.outB[(size_t)m * p.ldo + n] = (bf16_t)val;
                }
            }
        }
    }
}

// ---------------------------------------------------------------------------
// WMMA GEMM: C[M,N] = A[M,K] * B[N,K]^T  (B supplied as [N,K] so that the
// B-fragment (column-of-B) load is two contiguous 16B runs per lane).
// Block: 256 threads = 8 waves in a 2(M) x 4(N) grid.
// Block tile BM x BN, K step 32, double-buffered LDS tiles filled with
// global_load_async_to_lds_b128 (+8 bf16 row pad).
// Fragment layouts per CDNA5 ISA:
//   A 16x32: lane<16 -> K {0..7,16..23}, lane>=16 -> K {8..15,24..31}
//   B 32x16: lane<16 -> K {0..7,16..23}, lane>=16 -> K {8..15,24..31}
//   C 16x16: row = vgpr + 8*(lane>=16), col = lane&15
// ---------------------------------------------------------------------------
template <int BM, int BN, int MODE>
__global__ __launch_bounds__(256)
void gemm_bf16_wmma(GemmParams p)
{
    constexpr int BK  = 32;
    constexpr int PAD = 8;
    constexpr int LDT = BK + PAD;       // 40 bf16 = 80B row stride (16B aligned)
    constexpr int WM  = BM / 2;         // wave tile M
    constexpr int WN  = BN / 4;         // wave tile N
    constexpr int FM  = WM / 16;
    constexpr int FN  = WN / 16;
    constexpr int CA  = (BM * (BK / 8)) / 256;   // 16B chunks per thread (A)
    constexpr int CB  = (BN * (BK / 8)) / 256;   // 16B chunks per thread (B)
    static_assert(CA >= 1 && CB >= 1, "tile too small for 256 threads");

    __shared__ bf16_t As[2][BM][LDT];
    __shared__ bf16_t Bs[2][BN][LDT];

    const int tid   = threadIdx.x;
    const int wave  = tid >> 5;
    const int lane  = tid & 31;
    const int lhalf = lane >> 4;
    const int l15   = lane & 15;
    const int wm    = wave & 1;
    const int wn    = wave >> 1;

    const int m0 = blockIdx.y * BM;
    const int n0 = blockIdx.x * BN;
    const int z  = blockIdx.z;

    const bf16_t* Ag = p.A + (size_t)z * p.strideAz;
    const bf16_t* Bg = p.B + (size_t)z * p.strideBz;

    // Per-thread async-copy chunk descriptors (compile-time unrolled).
    const bf16_t* aSrc[CA];
    unsigned      aDst[2][CA];
#pragma unroll
    for (int j = 0; j < CA; ++j) {
        const int c   = tid + j * 256;
        const int r   = c >> 2;
        const int col = (c & 3) * 8;
        aSrc[j]    = Ag + (size_t)(m0 + r) * p.lda + col;
        aDst[0][j] = lds_offset_of(&As[0][r][col]);
        aDst[1][j] = lds_offset_of(&As[1][r][col]);
    }
    const bf16_t* bSrc[CB];
    unsigned      bDst[2][CB];
#pragma unroll
    for (int j = 0; j < CB; ++j) {
        const int c   = tid + j * 256;
        const int r   = c >> 2;
        const int col = (c & 3) * 8;
        bSrc[j]    = Bg + (size_t)(n0 + r) * p.ldb + col;
        bDst[0][j] = lds_offset_of(&Bs[0][r][col]);
        bDst[1][j] = lds_offset_of(&Bs[1][r][col]);
    }

    auto issue_tile = [&](int buf, int k0) {
#pragma unroll
        for (int j = 0; j < CA; ++j)
            async_g2l_b128(aDst[buf][j], aSrc[j] + k0);
#pragma unroll
        for (int j = 0; j < CB; ++j)
            async_g2l_b128(bDst[buf][j], bSrc[j] + k0);
    };

    v8f acc[FM][FN];
    const v8f vzero = {0.f, 0.f, 0.f, 0.f, 0.f, 0.f, 0.f, 0.f};
#pragma unroll
    for (int i = 0; i < FM; ++i)
#pragma unroll
        for (int j = 0; j < FN; ++j) acc[i][j] = vzero;

    // Prologue: stream first K-tile into buffer 0.
    issue_tile(0, 0);
    wait_asynccnt0();
    __syncthreads();

    int buf = 0;
    for (int k0 = 0; k0 < p.K; k0 += BK) {
        const bool more = (k0 + BK) < p.K;
        if (more) issue_tile(buf ^ 1, k0 + BK);   // overlap with compute

        v16bf aF[FM], bF[FN];
#pragma unroll
        for (int fm = 0; fm < FM; ++fm) {
            const int row = wm * WM + fm * 16 + l15;
            BfFrag f;
            f.u[0] = *(const uint4*)&As[buf][row][lhalf * 8];
            f.u[1] = *(const uint4*)&As[buf][row][lhalf * 8 + 16];
            aF[fm] = f.v;
        }
#pragma unroll
        for (int fn = 0; fn < FN; ++fn) {
            const int row = wn * WN + fn * 16 + l15;
            BfFrag f;
            f.u[0] = *(const uint4*)&Bs[buf][row][lhalf * 8];
            f.u[1] = *(const uint4*)&Bs[buf][row][lhalf * 8 + 16];
            bF[fn] = f.v;
        }
#pragma unroll
        for (int fm = 0; fm < FM; ++fm)
#pragma unroll
            for (int fn = 0; fn < FN; ++fn)
                acc[fm][fn] = __builtin_amdgcn_wmma_f32_16x16x32_bf16(
                    false, aF[fm], false, bF[fn], (short)0, acc[fm][fn],
                    false, false);

        if (more) wait_asynccnt0();   // next tile fully landed in LDS
        __syncthreads();              // publish + protect ping-pong reuse
        buf ^= 1;
    }

    gemm_epilogue<MODE, FM, FN>(p, acc, m0 + wm * WM, n0 + wn * WN,
                                lhalf, l15, z);
}

// ---------------------------------------------------------------------------
// Elementwise f32 -> bf16 convert.
// ---------------------------------------------------------------------------
__global__ __launch_bounds__(256)
void cvt_f32_to_bf16(const float* __restrict__ in, bf16_t* __restrict__ out,
                     int n)
{
    const int i = blockIdx.x * 256 + threadIdx.x;
    if (i < n) out[i] = (bf16_t)in[i];
}

// ---------------------------------------------------------------------------
// Transpose + convert: in f32 [K,N] -> out bf16 [N,K]. 32x32 LDS tile.
// ---------------------------------------------------------------------------
__global__ __launch_bounds__(256)
void transpose_cvt(const float* __restrict__ in, bf16_t* __restrict__ out,
                   int K, int N)
{
    __shared__ float tile[32][33];
    const int n0 = blockIdx.x * 32;
    const int k0 = blockIdx.y * 32;
    for (int i = threadIdx.y; i < 32; i += 8)
        tile[i][threadIdx.x] = in[(size_t)(k0 + i) * N + n0 + threadIdx.x];
    __syncthreads();
    for (int i = threadIdx.y; i < 32; i += 8)
        out[(size_t)(n0 + i) * K + k0 + threadIdx.x] =
            (bf16_t)tile[threadIdx.x][i];
}

// ---------------------------------------------------------------------------
// Wave32 + cross-wave reductions (256-thread block).
// ---------------------------------------------------------------------------
__device__ inline float block_reduce_max(float v, float* sbuf)
{
#pragma unroll
    for (int o = 16; o > 0; o >>= 1) v = fmaxf(v, __shfl_xor(v, o, 32));
    if ((threadIdx.x & 31) == 0) sbuf[threadIdx.x >> 5] = v;
    __syncthreads();
    float r = sbuf[0];
#pragma unroll
    for (int i = 1; i < 8; ++i) r = fmaxf(r, sbuf[i]);
    return r;
}

__device__ inline float block_reduce_sum(float v, float* sbuf)
{
#pragma unroll
    for (int o = 16; o > 0; o >>= 1) v += __shfl_xor(v, o, 32);
    if ((threadIdx.x & 31) == 0) sbuf[threadIdx.x >> 5] = v;
    __syncthreads();
    float r = sbuf[0];
#pragma unroll
    for (int i = 1; i < 8; ++i) r += sbuf[i];
    return r;
}

// ---------------------------------------------------------------------------
// In-place masked-score softmax over rows of 1024 (one block per row).
// Masked entries already hold exactly 0.0f (multiplicative mask, pre-softmax),
// so they contribute exp(0 - max) like the reference.
// ---------------------------------------------------------------------------
__global__ __launch_bounds__(256)
void softmax_rows_1024(bf16_t* __restrict__ P)
{
    __shared__ float s1[8], s2[8];
    bf16_t* row = P + (size_t)blockIdx.x * S_DIM;
    const int t = threadIdx.x;

    float v[4];
    float mx = -INFINITY;
#pragma unroll
    for (int i = 0; i < 4; ++i) {
        v[i] = (float)row[t + i * 256];
        mx = fmaxf(mx, v[i]);
    }
    mx = block_reduce_max(mx, s1);

    float sum = 0.f;
#pragma unroll
    for (int i = 0; i < 4; ++i) {
        v[i] = __expf(v[i] - mx);
        sum += v[i];
    }
    sum = block_reduce_sum(sum, s2);
    const float inv = 1.0f / sum;
#pragma unroll
    for (int i = 0; i < 4; ++i) row[t + i * 256] = (bf16_t)(v[i] * inv);
}

// ---------------------------------------------------------------------------
// out = LayerNorm(X + Y) * g + b over rows of 768 (one block per row).
// Optionally also emits a bf16 copy for the next WMMA GEMM.
// ---------------------------------------------------------------------------
__global__ __launch_bounds__(256)
void add_layernorm_768(const float* __restrict__ X, const float* __restrict__ Y,
                       const float* __restrict__ g, const float* __restrict__ b,
                       float* __restrict__ outF, bf16_t* __restrict__ outB)
{
    __shared__ float s1[8], s2[8];
    const size_t base = (size_t)blockIdx.x * E_DIM;
    const int t = threadIdx.x;

    float v[3];
    float sum = 0.f, sumsq = 0.f;
#pragma unroll
    for (int i = 0; i < 3; ++i) {
        const int idx = t + i * 256;
        v[i] = X[base + idx] + Y[base + idx];
        sum += v[i];
        sumsq += v[i] * v[i];
    }
    sum   = block_reduce_sum(sum, s1);
    __syncthreads();
    sumsq = block_reduce_sum(sumsq, s2);

    const float mean = sum * (1.0f / E_DIM);
    const float var  = sumsq * (1.0f / E_DIM) - mean * mean;
    const float rstd = rsqrtf(var + 1e-5f);

#pragma unroll
    for (int i = 0; i < 3; ++i) {
        const int idx = t + i * 256;
        const float o = (v[i] - mean) * rstd * g[idx] + b[idx];
        outF[base + idx] = o;
        if (outB) outB[base + idx] = (bf16_t)o;
    }
}

// ---------------------------------------------------------------------------
// Host-side orchestration.
// ---------------------------------------------------------------------------
extern "C" void kernel_launch(void* const* d_in, const int* in_sizes, int n_in,
                              void* d_out, int out_size, void* d_ws,
                              size_t ws_size, hipStream_t stream)
{
    (void)in_sizes; (void)n_in; (void)out_size; (void)ws_size;

    const float* x   = (const float*)d_in[0];
    // d_in[1] (smask) is recomputed analytically: mask[q,k] = ((q&k)==0).
    const float* Wq  = (const float*)d_in[2];
    const float* bq  = (const float*)d_in[3];
    const float* Wk  = (const float*)d_in[4];
    const float* bk  = (const float*)d_in[5];
    const float* Wv  = (const float*)d_in[6];
    const float* bv  = (const float*)d_in[7];
    const float* Wo  = (const float*)d_in[8];
    const float* bo  = (const float*)d_in[9];
    const float* W1  = (const float*)d_in[10];
    const float* b1  = (const float*)d_in[11];
    const float* W2  = (const float*)d_in[12];
    const float* b2  = (const float*)d_in[13];
    const float* g1  = (const float*)d_in[14];
    const float* be1 = (const float*)d_in[15];
    const float* g2  = (const float*)d_in[16];
    const float* be2 = (const float*)d_in[17];
    float* out = (float*)d_out;

    char* ws = (char*)d_ws;
    size_t off = 0;
    auto alloc = [&](size_t bytes) -> char* {
        char* p = ws + off;
        off += (bytes + 255) & ~(size_t)255;
        return p;
    };

    const size_t M = (size_t)B_DIM * S_DIM;              // 4096 rows
    bf16_t* Xb   = (bf16_t*)alloc(M * E_DIM * 2);
    bf16_t* WqT  = (bf16_t*)alloc((size_t)E_DIM * E_DIM * 2);
    bf16_t* WkT  = (bf16_t*)alloc((size_t)E_DIM * E_DIM * 2);
    bf16_t* WvT  = (bf16_t*)alloc((size_t)E_DIM * E_DIM * 2);
    bf16_t* WoT  = (bf16_t*)alloc((size_t)E_DIM * E_DIM * 2);
    bf16_t* W1T  = (bf16_t*)alloc((size_t)F_DIM * E_DIM * 2);
    bf16_t* W2T  = (bf16_t*)alloc((size_t)E_DIM * F_DIM * 2);
    bf16_t* Qb   = (bf16_t*)alloc((size_t)B_DIM * H_DIM * S_DIM * D_DIM * 2);
    bf16_t* Kb   = (bf16_t*)alloc((size_t)B_DIM * H_DIM * S_DIM * D_DIM * 2);
    bf16_t* VT   = (bf16_t*)alloc((size_t)B_DIM * H_DIM * D_DIM * S_DIM * 2);
    bf16_t* P    = (bf16_t*)alloc((size_t)B_DIM * H_DIM * S_DIM * S_DIM * 2);
    bf16_t* Att  = (bf16_t*)alloc(M * E_DIM * 2);
    float*  AttO = (float*) alloc(M * E_DIM * 4);
    float*  Ln1f = (float*) alloc(M * E_DIM * 4);
    bf16_t* Ln1b = (bf16_t*)alloc(M * E_DIM * 2);
    float*  H2v  = (float*) alloc(M * E_DIM * 4);
    bf16_t* H1   = (bf16_t*)alloc(M * F_DIM * 2);

    // 1) x -> bf16
    {
        const int n = (int)(M * E_DIM);
        cvt_f32_to_bf16<<<(n + 255) / 256, 256, 0, stream>>>(x, Xb, n);
    }
    // 2) weights -> bf16 transposed [N,K]
    transpose_cvt<<<dim3(24, 24), dim3(32, 8), 0, stream>>>(Wq, WqT, E_DIM, E_DIM);
    transpose_cvt<<<dim3(24, 24), dim3(32, 8), 0, stream>>>(Wk, WkT, E_DIM, E_DIM);
    transpose_cvt<<<dim3(24, 24), dim3(32, 8), 0, stream>>>(Wv, WvT, E_DIM, E_DIM);
    transpose_cvt<<<dim3(24, 24), dim3(32, 8), 0, stream>>>(Wo, WoT, E_DIM, E_DIM);
    transpose_cvt<<<dim3(96, 24), dim3(32, 8), 0, stream>>>(W1, W1T, E_DIM, F_DIM);
    transpose_cvt<<<dim3(24, 96), dim3(32, 8), 0, stream>>>(W2, W2T, F_DIM, E_DIM);

    GemmParams p{};

    // 3) Q/K projections -> [B,H,S,D] bf16 ; V -> [B,H,D,S] bf16
    p.A = Xb; p.lda = E_DIM; p.ldb = E_DIM;
    p.M = (int)M; p.N = E_DIM; p.K = E_DIM;
    p.strideAz = p.strideBz = p.strideOz = 0; p.scale = 1.f;
    p.B = WqT; p.bias = bq; p.outB = Qb;
    gemm_bf16_wmma<128, 128, 1><<<dim3(6, 32, 1), 256, 0, stream>>>(p);
    p.B = WkT; p.bias = bk; p.outB = Kb;
    gemm_bf16_wmma<128, 128, 1><<<dim3(6, 32, 1), 256, 0, stream>>>(p);
    p.B = WvT; p.bias = bv; p.outB = VT;
    gemm_bf16_wmma<128, 128, 2><<<dim3(6, 32, 1), 256, 0, stream>>>(p);

    // 4) scores[z] = (Q_z * K_z^T) * 1/sqrt(D), Sierpinski-masked, bf16
    p = GemmParams{};
    p.A = Qb; p.B = Kb; p.outB = P; p.bias = nullptr;
    p.M = S_DIM; p.N = S_DIM; p.K = D_DIM;
    p.lda = D_DIM; p.ldb = D_DIM; p.ldo = S_DIM;
    p.strideAz = (long long)S_DIM * D_DIM;
    p.strideBz = (long long)S_DIM * D_DIM;
    p.strideOz = (long long)S_DIM * S_DIM;
    p.scale = 0.125f;                         // 1/sqrt(64)
    gemm_bf16_wmma<128, 128, 3><<<dim3(8, 8, B_DIM * H_DIM), 256, 0, stream>>>(p);

    // 5) softmax in place over 1024-wide rows
    softmax_rows_1024<<<B_DIM * H_DIM * S_DIM, 256, 0, stream>>>(P);

    // 6) attended[z] = P_z * V_z, scattered back to [B,S,E]
    p = GemmParams{};
    p.A = P; p.B = VT; p.outB = Att; p.bias = nullptr;
    p.M = S_DIM; p.N = D_DIM; p.K = S_DIM;
    p.lda = S_DIM; p.ldb = S_DIM; p.ldo = 0;
    p.strideAz = (long long)S_DIM * S_DIM;
    p.strideBz = (long long)D_DIM * S_DIM;
    p.strideOz = 0; p.scale = 1.f;
    gemm_bf16_wmma<128, 64, 4><<<dim3(1, 8, B_DIM * H_DIM), 256, 0, stream>>>(p);

    // 7) attn_out = Att * Wo^T + bo  (f32)
    p = GemmParams{};
    p.A = Att; p.B = WoT; p.outF = AttO; p.bias = bo;
    p.M = (int)M; p.N = E_DIM; p.K = E_DIM;
    p.lda = E_DIM; p.ldb = E_DIM; p.ldo = E_DIM; p.scale = 1.f;
    gemm_bf16_wmma<128, 128, 0><<<dim3(6, 32, 1), 256, 0, stream>>>(p);

    // 8) LN1(x + attn_out) -> f32 + bf16
    add_layernorm_768<<<(int)M, 256, 0, stream>>>(x, AttO, g1, be1, Ln1f, Ln1b);

    // 9) h1 = GELU(Ln1 * W1^T + b1)  bf16
    p = GemmParams{};
    p.A = Ln1b; p.B = W1T; p.outB = H1; p.bias = b1;
    p.M = (int)M; p.N = F_DIM; p.K = E_DIM;
    p.lda = E_DIM; p.ldb = E_DIM; p.ldo = F_DIM; p.scale = 1.f;
    gemm_bf16_wmma<128, 128, 5><<<dim3(24, 32, 1), 256, 0, stream>>>(p);

    // 10) h2 = h1 * W2^T + b2  (f32)
    p = GemmParams{};
    p.A = H1; p.B = W2T; p.outF = H2v; p.bias = b2;
    p.M = (int)M; p.N = E_DIM; p.K = F_DIM;
    p.lda = F_DIM; p.ldb = F_DIM; p.ldo = E_DIM; p.scale = 1.f;
    gemm_bf16_wmma<128, 128, 0><<<dim3(6, 32, 1), 256, 0, stream>>>(p);

    // 11) out = LN2(Ln1 + h2)
    add_layernorm_768<<<(int)M, 256, 0, stream>>>(Ln1f, H2v, g2, be2, out,
                                                  nullptr);
}
